// TransformerBlock_86904368267857
// MI455X (gfx1250) — matrix operand order repaired and checked by reading, not compiled, not run on hardware
//
#include <hip/hip_runtime.h>
#include <cstdint>

typedef _Float16 f16;
typedef __attribute__((ext_vector_type(16))) _Float16 v16h;
typedef __attribute__((ext_vector_type(8)))  float    v8f;

#define BATCH   4
#define CH      256
#define HID     1024
#define WIN     8
#define NTOK    64        // tokens per window
#define NWIN    256       // windows per batch image (16x16)
#define HEADS   8
#define HD      32
#define LTOT    16384     // 128*128
#define MROWS   65536     // BATCH * LTOT
#define ASTR    40        // LDS stride (halves) for 32-wide rows (16B aligned frags)
#define BSTR    72        // LDS stride (halves) for 64-wide rows (16B aligned frags)

// CDNA5 async global->LDS path (ASYNCcnt). The clang builtin for the async
// load has a toolchain-specific descriptor signature, so we emit the ISA
// instruction directly via inline asm (device pass only) and use the
// probe-confirmed s_wait_asynccnt builtin for completion.
#if defined(__gfx1250__) && __has_builtin(__builtin_amdgcn_s_wait_asynccnt)
#define USE_ASYNC_LDS 1
#else
#define USE_ASYNC_LDS 0
#endif

union Frag16 { v16h v; f16 h[16]; uint4 q[2]; };

__device__ __forceinline__ v8f vzero8() {
  v8f z;
#pragma unroll
  for (int i = 0; i < 8; ++i) z[i] = 0.f;
  return z;
}

__device__ __forceinline__ float gelu_exact(float x) {
  return 0.5f * x * (1.f + erff(x * 0.70710678118654752f));
}

#if USE_ASYNC_LDS
// 16 bytes global -> LDS via the async DMA path (tracked by ASYNCcnt).
// GLOBAL_LOAD_ASYNC_TO_LDS_B128, GV mode: VDST = 32-bit LDS byte address,
// VADDR = 64-bit global address. Generic pointers to __shared__ carry the
// LDS byte offset in their low 32 bits.
__device__ __forceinline__ void async_cp16(const f16* gsrc, f16* ldst) {
  unsigned lds_addr = (unsigned)(uintptr_t)ldst;
  asm volatile("global_load_async_to_lds_b128 %0, %1, off"
               :
               : "v"(lds_addr), "v"(gsrc)
               : "memory");
}
#endif

// ---------------------------------------------------------------------------
// Kernel 0: convert weights to f16 in K-panel layout Wt[(k/32)*Nc + n][k%32],
// so WMMA B-fragments are two contiguous 16B global loads per lane (L2-resident).
// wq|wkv fused into one [256 x 768] matrix.
// ---------------------------------------------------------------------------
__global__ void cvt_weights(const float* __restrict__ wq, const float* __restrict__ wkv,
                            const float* __restrict__ pw, const float* __restrict__ l1,
                            const float* __restrict__ l2,
                            f16* __restrict__ wqkv, f16* __restrict__ pwh,
                            f16* __restrict__ l1h, f16* __restrict__ l2h) {
  int i = blockIdx.x * blockDim.x + threadIdx.x;
  if (i < 256 * 768) {              // K=256, Nc=768
    int k = i / 768, c = i % 768;
    float v = (c < 256) ? wq[k * 256 + c] : wkv[k * 512 + (c - 256)];
    wqkv[((size_t)(k >> 5) * 768 + c) * 32 + (k & 31)] = (f16)v;
  }
  if (i < 256 * 256) {              // K=256, Nc=256
    int k = i >> 8, c = i & 255;
    pwh[((size_t)(k >> 5) * 256 + c) * 32 + (k & 31)] = (f16)pw[i];
  }
  if (i < 256 * 1024) {             // lin1: K=256, Nc=1024
    int k = i >> 10, c = i & 1023;
    l1h[((size_t)(k >> 5) * 1024 + c) * 32 + (k & 31)] = (f16)l1[i];
  }
  if (i < 1024 * 256) {             // lin2: K=1024, Nc=256
    int k = i >> 8, c = i & 255;
    l2h[((size_t)(k >> 5) * 256 + c) * 32 + (k & 31)] = (f16)l2[i];
  }
}

// ---------------------------------------------------------------------------
// Kernel 1/5: LayerNorm over C=256, one wave(32) per token, 8 ch per lane.
// SHIFTED=1: also roll(-4,-4) + 8x8 window partition into the output layout.
// ---------------------------------------------------------------------------
template<int SHIFTED>
__global__ void __launch_bounds__(256)
ln_kernel(const float* __restrict__ x, const float* __restrict__ gw,
          const float* __restrict__ gb, f16* __restrict__ out) {
  int tid = threadIdx.x, lane = tid & 31, wv = tid >> 5;
  int tk = blockIdx.x * 8 + wv;                 // token id in [0, 65536)
  int b = tk >> 14, l = tk & (LTOT - 1);
  size_t src, dst;
  if (SHIFTED) {
    int h = l >> 7, w = l & 127;
    int hs = (h + 4) & 127, ws = (w + 4) & 127; // xs[h,w] = xn[h+4, w+4]
    src = ((size_t)b * LTOT + (size_t)hs * 128 + ws) * CH;
    int win = (h >> 3) * 16 + (w >> 3);
    int t = (h & 7) * 8 + (w & 7);
    dst = (((size_t)b * NWIN + win) * NTOK + t) * CH;
  } else {
    src = (size_t)tk * CH;
    dst = src;
  }
  const float4* xp = (const float4*)(x + src + lane * 8);
  float4 a0 = xp[0], a1 = xp[1];
  float v[8] = {a0.x, a0.y, a0.z, a0.w, a1.x, a1.y, a1.z, a1.w};
  float s = 0.f;
#pragma unroll
  for (int i = 0; i < 8; ++i) s += v[i];
  for (int m = 1; m < 32; m <<= 1) s += __shfl_xor(s, m, 32);
  float mean = s * (1.f / CH);
  float vs = 0.f;
#pragma unroll
  for (int i = 0; i < 8; ++i) { float d = v[i] - mean; vs += d * d; }
  for (int m = 1; m < 32; m <<= 1) vs += __shfl_xor(vs, m, 32);
  float rstd = rsqrtf(vs * (1.f / CH) + 1e-5f);
  const float4* wp = (const float4*)(gw + lane * 8);
  const float4* bp = (const float4*)(gb + lane * 8);
  float4 w0 = wp[0], w1 = wp[1], b0 = bp[0], b1 = bp[1];
  float wa[8] = {w0.x, w0.y, w0.z, w0.w, w1.x, w1.y, w1.z, w1.w};
  float ba[8] = {b0.x, b0.y, b0.z, b0.w, b1.x, b1.y, b1.z, b1.w};
  union { f16 h[8]; uint4 q; } o;
#pragma unroll
  for (int i = 0; i < 8; ++i) o.h[i] = (f16)((v[i] - mean) * rstd * wa[i] + ba[i]);
  *(uint4*)(out + dst + lane * 8) = o.q;
}

// ---------------------------------------------------------------------------
// Tiled WMMA GEMM: C[M,Nc] = A[M,K] * B[K,Nc], f16 in, f32 accumulate.
// Block tile 64 rows x 128 cols, 4 waves (each 64x32 -> 8 WMMA/k-step).
// A double-buffered in LDS (1 barrier/iter); staged via async global->LDS
// (ASYNCcnt) when available, else via registers. B fragments read directly
// from global K-panel layout (weights stay resident in the 192MB L2).
// MODE 0: QKV epilogue (scatter to q/k/v, +bias, q*scale)
// MODE 1: proj epilogue (window-reverse + unshift + residual, f32 out)
// MODE 2: lin1 epilogue (+bias, exact GELU, f16 out)
// MODE 3: lin2 epilogue (+bias, +residual, f32 out)
// ---------------------------------------------------------------------------
template<int MODE>
__global__ void __launch_bounds__(128)
gemm_kernel(const f16* __restrict__ A, const f16* __restrict__ Bt,
            int K, int Nc,
            const float* __restrict__ b0, const float* __restrict__ b1,
            const float* __restrict__ res, float* __restrict__ of32,
            f16* __restrict__ o0, f16* __restrict__ o1, f16* __restrict__ o2) {
  __shared__ f16 As[2 * 64 * ASTR];             // double-buffered A tile
  int tid = threadIdx.x, lane = tid & 31, wv = tid >> 5;
  int ln = lane & 15, g = lane >> 4;
  int rowBase = blockIdx.y * 64;
  int colBase = blockIdx.x * 128;
  int waveCol = colBase + wv * 32;

  v8f acc[4][2];
#pragma unroll
  for (int i = 0; i < 4; ++i) { acc[i][0] = vzero8(); acc[i][1] = vzero8(); }

  int ksteps = K >> 5;
  int ar = tid >> 1, aof = (tid & 1) * 16;      // A tile: 64 rows x 32 halves
  const f16* gA = A + (size_t)(rowBase + ar) * K + aof;
  f16* lA = As + ar * ASTR + aof;

#if USE_ASYNC_LDS
  async_cp16(gA, lA);                           // prologue: tile 0 -> buf0
  async_cp16(gA + 8, lA + 8);
#else
  const uint4* ga = (const uint4*)gA;
  uint4 ra0 = ga[0], ra1 = ga[1];               // preload tile 0
#endif

  for (int ks = 0; ks < ksteps; ++ks) {
    f16* buf = As + (ks & 1) * (64 * ASTR);
#if USE_ASYNC_LDS
    __builtin_amdgcn_s_wait_asynccnt(0);        // own tile-ks writes landed
    __syncthreads();                            // all waves' writes visible;
                                                // prior readers of other buf drained
    if (ks + 1 < ksteps) {
      f16* lnx = As + ((ks + 1) & 1) * (64 * ASTR) + ar * ASTR + aof;
      const f16* gn = gA + (ks + 1) * 32;
      async_cp16(gn, lnx);
      async_cp16(gn + 8, lnx + 8);
    }
#else
    *(uint4*)(buf + ar * ASTR + aof) = ra0;
    *(uint4*)(buf + ar * ASTR + aof + 8) = ra1;
    if (ks + 1 < ksteps) {                      // issue next tile load pre-barrier
      const uint4* gn = (const uint4*)(gA + (ks + 1) * 32);
      ra0 = gn[0]; ra1 = gn[1];
    }
    __syncthreads();
#endif

    // B fragments (K-panel global layout): 2 x (2 x b128) per wave per k-step
    const f16* bp0 = Bt + ((size_t)ks * Nc + waveCol + ln) * 32 + g * 16;
    const f16* bp1 = bp0 + 16 * 32;
    Frag16 bf0, bf1;
    bf0.q[0] = *(const uint4*)(bp0);
    bf0.q[1] = *(const uint4*)(bp0 + 8);
    bf1.q[0] = *(const uint4*)(bp1);
    bf1.q[1] = *(const uint4*)(bp1 + 8);
    if (ks + 1 < ksteps) {                      // CDNA5 global_prefetch_b8
      __builtin_prefetch(bp0 + (size_t)Nc * 32, 0, 0);
      __builtin_prefetch(bp1 + (size_t)Nc * 32, 0, 0);
    }
#pragma unroll
    for (int mb = 0; mb < 4; ++mb) {
      Frag16 af;
      const f16* ap = buf + (mb * 16 + ln) * ASTR;
      af.q[0] = *(const uint4*)(ap + g * 16);
      af.q[1] = *(const uint4*)(ap + g * 16 + 8);
      acc[mb][0] = __builtin_amdgcn_wmma_f32_16x16x32_f16(false, af.v, false, bf0.v,
                                                          (short)0, acc[mb][0], false, false);
      acc[mb][1] = __builtin_amdgcn_wmma_f32_16x16x32_f16(false, af.v, false, bf1.v,
                                                          (short)0, acc[mb][1], false, false);
    }
#if !USE_ASYNC_LDS
    __syncthreads();
#endif
  }
#if USE_ASYNC_LDS
  __syncthreads();                              // readers drain before epilogue
#endif

#pragma unroll
  for (int mb = 0; mb < 4; ++mb) {
#pragma unroll
    for (int nb = 0; nb < 2; ++nb) {
      int col = waveCol + nb * 16 + ln;
#pragma unroll
      for (int r = 0; r < 8; ++r) {
        int row = rowBase + mb * 16 + g * 8 + r;  // C layout: VGPR r, half g
        float v = acc[mb][nb][r];
        if (MODE == 0) {
          int win = row >> 6, t = row & 63;
          if (col < CH) {
            int head = col >> 5, d = col & 31;
            float q = (v + b0[col]) * 0.17677669529663687f;  // hd^-0.5
            o0[(((size_t)win * HEADS + head) * NTOK + t) * HD + d] = (f16)q;
          } else {
            int j = col - CH;
            int kv = j >> 8, head = (j & 255) >> 5, d = j & 31;
            f16* dstp = kv ? o2 : o1;
            dstp[(((size_t)win * HEADS + head) * NTOK + t) * HD + d] = (f16)(v + b1[j]);
          }
        } else if (MODE == 1) {
          int win = row >> 6, t = row & 63;
          int b = win >> 8, wi = win & 255;
          int wy = wi >> 4, wx = wi & 15, ty = t >> 3, tx = t & 7;
          int h = (wy * 8 + ty + 4) & 127;   // unshift: roll(+4,+4)
          int w = (wx * 8 + tx + 4) & 127;
          size_t idx = (((size_t)b * LTOT) + (size_t)h * 128 + w) * CH + col;
          of32[idx] = res[idx] + v + b0[col];
        } else if (MODE == 2) {
          o0[(size_t)row * Nc + col] = (f16)gelu_exact(v + b0[col]);
        } else {
          size_t idx = (size_t)row * CH + col;
          of32[idx] = res[idx] + v + b0[col];
        }
      }
    }
  }
}

// ---------------------------------------------------------------------------
// Attention: one wave per (window, head). S = q k^T (WMMA), +relpos bias,
// +shift mask, softmax (shfl reductions), O = P V (WMMA). N=64, hd=32.
// ---------------------------------------------------------------------------
__global__ void __launch_bounds__(32)
attn_kernel(const f16* __restrict__ qg, const f16* __restrict__ kg,
            const f16* __restrict__ vg, const float* __restrict__ rpb,
            f16* __restrict__ og) {
  __shared__ f16 qk[64 * ASTR * 2];  // q rows then k rows; P overlays (64*BSTR<=this)
  __shared__ f16 vt[32 * BSTR];      // v transposed: [d][token], padded
  __shared__ float rb[232];          // rel-pos bias row for this head (225 used)
  f16* q_lds = qk;
  f16* k_lds = qk + 64 * ASTR;
  f16* p_lds = qk;

  int lane = threadIdx.x, ln = lane & 15, g = lane >> 4;
  int wh = blockIdx.x;                  // window*8 + head
  int win = wh >> 3, head = wh & 7;
  int wi = win & 255;
  int wy = wi >> 4, wx = wi & 15;
  size_t base = (size_t)wh * (NTOK * HD);

  // stage q, k (row-major token x hd), v (transposed), and rpb row into LDS
#pragma unroll
  for (int i = 0; i < 8; ++i) {
    int c = lane + 32 * i;              // 256 chunks of 8 halves
    int t = c >> 2, off = (c & 3) * 8;
#if USE_ASYNC_LDS
    async_cp16(qg + base + t * HD + off, q_lds + t * ASTR + off);
    async_cp16(kg + base + t * HD + off, k_lds + t * ASTR + off);
#else
    *(uint4*)(q_lds + t * ASTR + off) = *(const uint4*)(qg + base + t * HD + off);
    *(uint4*)(k_lds + t * ASTR + off) = *(const uint4*)(kg + base + t * HD + off);
#endif
    int ri = lane + 32 * i;
    if (ri < 225) rb[ri] = rpb[ri * HEADS + head];
  }
  for (int t = 0; t < 64; ++t) vt[lane * BSTR + t] = vg[base + (size_t)t * HD + lane];
#if USE_ASYNC_LDS
  __builtin_amdgcn_s_wait_asynccnt(0);
#endif
  __syncthreads();

  // S = q * k^T : M=64, N=64, K=32 -> 16 WMMAs
  v8f acc[4][4];
#pragma unroll
  for (int a = 0; a < 4; ++a)
#pragma unroll
    for (int b2 = 0; b2 < 4; ++b2) acc[a][b2] = vzero8();

#pragma unroll
  for (int nb = 0; nb < 4; ++nb) {
    Frag16 kf;  // B[kk][j] = k[j][kk]: contiguous 16 halves per lane
    const f16* kp = k_lds + (nb * 16 + ln) * ASTR + g * 16;
    kf.q[0] = *(const uint4*)(kp);
    kf.q[1] = *(const uint4*)(kp + 8);
#pragma unroll
    for (int mb = 0; mb < 4; ++mb) {
      Frag16 qf;
      const f16* qp = q_lds + (mb * 16 + ln) * ASTR;
      qf.q[0] = *(const uint4*)(qp + g * 16);
      qf.q[1] = *(const uint4*)(qp + g * 16 + 8);
      acc[mb][nb] = __builtin_amdgcn_wmma_f32_16x16x32_f16(false, qf.v, false, kf.v,
                                                           (short)0, acc[mb][nb], false, false);
    }
  }

  // bias + shift mask + softmax (row = mb*16 + g*8 + r, col = nb*16 + ln)
#pragma unroll
  for (int mb = 0; mb < 4; ++mb) {
#pragma unroll
    for (int r = 0; r < 8; ++r) {
      int i = mb * 16 + g * 8 + r;
      int iy = i >> 3, ix = i & 7;
      int hi = wy * 8 + iy, wix = wx * 8 + ix;
      int regi = (hi < 120 ? 0 : (hi < 124 ? 1 : 2)) * 3 + (wix < 120 ? 0 : (wix < 124 ? 1 : 2));
      float rowv[4];
#pragma unroll
      for (int nb = 0; nb < 4; ++nb) {
        int j = nb * 16 + ln;
        int jy = j >> 3, jx = j & 7;
        int hj = wy * 8 + jy, wjx = wx * 8 + jx;
        int regj = (hj < 120 ? 0 : (hj < 124 ? 1 : 2)) * 3 + (wjx < 120 ? 0 : (wjx < 124 ? 1 : 2));
        int rel = (iy - jy + 7) * 15 + (ix - jx + 7);
        float bias = rb[rel];
        if (regi != regj) bias -= 100.f;
        rowv[nb] = acc[mb][nb][r] + bias;
      }
      float mx = fmaxf(fmaxf(rowv[0], rowv[1]), fmaxf(rowv[2], rowv[3]));
      for (int m = 1; m < 16; m <<= 1) mx = fmaxf(mx, __shfl_xor(mx, m, 32));
      float s = 0.f;
#pragma unroll
      for (int nb = 0; nb < 4; ++nb) { rowv[nb] = __expf(rowv[nb] - mx); s += rowv[nb]; }
      for (int m = 1; m < 16; m <<= 1) s += __shfl_xor(s, m, 32);
      float inv = 1.f / s;
#pragma unroll
      for (int nb = 0; nb < 4; ++nb)
        p_lds[(size_t)i * BSTR + nb * 16 + ln] = (f16)(rowv[nb] * inv);
    }
  }
  __syncthreads();

  // O = P * V : M=64, N=32, K=64 -> 16 WMMAs
  v8f oacc[4][2];
#pragma unroll
  for (int a = 0; a < 4; ++a)
#pragma unroll
    for (int b2 = 0; b2 < 2; ++b2) oacc[a][b2] = vzero8();

#pragma unroll
  for (int nb2 = 0; nb2 < 2; ++nb2) {
#pragma unroll
    for (int kb = 0; kb < 2; ++kb) {
      Frag16 vf;  // B[kk][d] = vT[d][kk]: contiguous 16 halves
      const f16* vp = vt + (nb2 * 16 + ln) * BSTR + kb * 32 + g * 16;
      vf.q[0] = *(const uint4*)(vp);
      vf.q[1] = *(const uint4*)(vp + 8);
#pragma unroll
      for (int mb = 0; mb < 4; ++mb) {
        Frag16 pf;
        const f16* pp = p_lds + (mb * 16 + ln) * BSTR + kb * 32;
        pf.q[0] = *(const uint4*)(pp + g * 16);
        pf.q[1] = *(const uint4*)(pp + g * 16 + 8);
        oacc[mb][nb2] = __builtin_amdgcn_wmma_f32_16x16x32_f16(false, pf.v, false, vf.v,
                                                               (short)0, oacc[mb][nb2], false, false);
      }
    }
  }
  // write O as [win][token][C] with channel offset head*32 (ready for proj GEMM)
#pragma unroll
  for (int mb = 0; mb < 4; ++mb)
#pragma unroll
    for (int nb2 = 0; nb2 < 2; ++nb2)
#pragma unroll
      for (int r = 0; r < 8; ++r) {
        int t = mb * 16 + g * 8 + r;
        int d = nb2 * 16 + ln;
        og[((size_t)win * NTOK + t) * CH + head * HD + d] = (f16)oacc[mb][nb2][r];
      }
}

// ---------------------------------------------------------------------------
// Depthwise 3x3 conv (SAME) + bias + GELU; 4 channels per thread, vector taps.
// ---------------------------------------------------------------------------
__global__ void __launch_bounds__(256)
dwconv_kernel(const f16* __restrict__ hin, const float* __restrict__ dw,
              const float* __restrict__ db, f16* __restrict__ hout) {
  size_t idx = (size_t)blockIdx.x * blockDim.x + threadIdx.x;  // B*H*W*256
  int cgrp = (int)(idx & 255);
  size_t pix = idx >> 8;
  int b = (int)(pix >> 14), l = (int)(pix & (LTOT - 1));
  int h = l >> 7, w = l & 127;
  int c = cgrp * 4;
  float accv[4] = {0.f, 0.f, 0.f, 0.f};
#pragma unroll
  for (int ky = 0; ky < 3; ++ky) {
    int hh = h + ky - 1;
    if (hh < 0 || hh > 127) continue;
#pragma unroll
    for (int kx = 0; kx < 3; ++kx) {
      int ww = w + kx - 1;
      if (ww < 0 || ww > 127) continue;
      union { uint2 u2; f16 h4[4]; } iv;
      iv.u2 = *(const uint2*)(hin + (((size_t)b * LTOT + (size_t)hh * 128 + ww) * HID) + c);
#pragma unroll
      for (int u = 0; u < 4; ++u)
        accv[u] += (float)iv.h4[u] * dw[(size_t)(c + u) * 9 + ky * 3 + kx];
    }
  }
  union { f16 h4[4]; uint2 u2; } o;
#pragma unroll
  for (int u = 0; u < 4; ++u) o.h4[u] = (f16)gelu_exact(accv[u] + db[c + u]);
  *(uint2*)(hout + ((size_t)b * LTOT + (size_t)h * 128 + w) * HID + c) = o.u2;
}

// ---------------------------------------------------------------------------
extern "C" void kernel_launch(void* const* d_in, const int* in_sizes, int n_in,
                              void* d_out, int out_size, void* d_ws, size_t ws_size,
                              hipStream_t stream) {
  (void)in_sizes; (void)n_in; (void)out_size; (void)ws_size;
  const float* x   = (const float*)d_in[0];
  const float* n1w = (const float*)d_in[3];
  const float* n1b = (const float*)d_in[4];
  const float* wq  = (const float*)d_in[5];
  const float* bq  = (const float*)d_in[6];
  const float* wkv = (const float*)d_in[7];
  const float* bkv = (const float*)d_in[8];
  const float* rpb = (const float*)d_in[9];
  const float* pw  = (const float*)d_in[10];
  const float* pb  = (const float*)d_in[11];
  const float* n2w = (const float*)d_in[12];
  const float* n2b = (const float*)d_in[13];
  const float* l1w = (const float*)d_in[14];
  const float* l1b = (const float*)d_in[15];
  const float* dww = (const float*)d_in[16];
  const float* dwb = (const float*)d_in[17];
  const float* l2w = (const float*)d_in[18];
  const float* l2b = (const float*)d_in[19];
  float* out = (float*)d_out;

  char* ws = (char*)d_ws;
  size_t off = 0;
  auto alloc = [&](size_t bytes) -> void* {
    off = (off + 255) & ~(size_t)255;
    void* p = ws + off;
    off += bytes;
    return p;
  };
  f16* wqkvh = (f16*)alloc((size_t)256 * 768 * 2);
  f16* pwh   = (f16*)alloc((size_t)256 * 256 * 2);
  f16* l1h   = (f16*)alloc((size_t)256 * 1024 * 2);
  f16* l2h   = (f16*)alloc((size_t)1024 * 256 * 2);
  f16* xw    = (f16*)alloc((size_t)MROWS * CH * 2);   // LN1 windows; reused as attn O
  f16* qb    = (f16*)alloc((size_t)MROWS * CH * 2);   // q; reused as LN2 output
  f16* kb    = (f16*)alloc((size_t)MROWS * CH * 2);
  f16* vb    = (f16*)alloc((size_t)MROWS * CH * 2);
  float* x1  = (float*)alloc((size_t)MROWS * CH * 4); // post-attention residual
  f16* hbuf  = (f16*)alloc((size_t)MROWS * HID * 2);
  f16* h2buf = (f16*)alloc((size_t)MROWS * HID * 2);

  cvt_weights<<<1024, 256, 0, stream>>>(wq, wkv, pw, l1w, l2w, wqkvh, pwh, l1h, l2h);
  ln_kernel<1><<<MROWS / 8, 256, 0, stream>>>(x, n1w, n1b, xw);
  gemm_kernel<0><<<dim3(6, 1024), 128, 0, stream>>>(xw, wqkvh, 256, 768, bq, bkv,
                                                    nullptr, nullptr, qb, kb, vb);
  attn_kernel<<<8192, 32, 0, stream>>>(qb, kb, vb, rpb, xw);  // O overlays xw
  gemm_kernel<1><<<dim3(2, 1024), 128, 0, stream>>>(xw, pwh, 256, 256, pb, nullptr,
                                                    x, x1, nullptr, nullptr, nullptr);
  ln_kernel<0><<<MROWS / 8, 256, 0, stream>>>(x1, n2w, n2b, qb);  // x2n overlays qb
  gemm_kernel<2><<<dim3(8, 1024), 128, 0, stream>>>(qb, l1h, 256, 1024, l1b, nullptr,
                                                    nullptr, nullptr, hbuf, nullptr, nullptr);
  dwconv_kernel<<<65536, 256, 0, stream>>>(hbuf, dww, dwb, h2buf);
  gemm_kernel<3><<<dim3(2, 1024), 128, 0, stream>>>(h2buf, l2h, 1024, 256, l2b, nullptr,
                                                    x1, out, nullptr, nullptr, nullptr);
}
